// ModelNer_16810501997104
// MI455X (gfx1250) — compile-verified, matrix operand order checked by reading, not run on hardware
//
#include <hip/hip_runtime.h>
#include <math.h>

// Problem constants (from reference)
#define TT 512
#define BB 64
#define EE 256
#define HH 256
#define NTAG 49

typedef __attribute__((ext_vector_type(16))) __bf16 v16bf;
typedef __attribute__((ext_vector_type(8)))  __bf16 v8bf;
typedef __attribute__((ext_vector_type(8)))  float  v8f;

#define AS3 __attribute__((address_space(3)))

__device__ __forceinline__ __bf16 f2bf(float f) {
    union { float f; unsigned u; } v; v.f = f;
    unsigned r = (v.u + 0x7FFFu + ((v.u >> 16) & 1u)) >> 16;
    unsigned short h = (unsigned short)r;
    __bf16 out; __builtin_memcpy(&out, &h, 2); return out;
}
__device__ __forceinline__ float bf2f(__bf16 b) {
    unsigned short h; __builtin_memcpy(&h, &b, 2);
    unsigned u = ((unsigned)h) << 16;
    float f; __builtin_memcpy(&f, &u, 4); return f;
}
__device__ __forceinline__ float sigf(float x) { return 1.0f / (1.0f + expf(-x)); }

// LDS byte offset (relative to wave LDS base) of a shared-memory pointer.
__device__ __forceinline__ unsigned lds_off_of(void* p) {
    return (unsigned)(unsigned long long)(AS3 char*)p;
}

// CDNA5 async global->LDS copy (ASYNCcnt-tracked), 16B per lane.
__device__ __forceinline__ void async_copy_b128(unsigned lds_byte_off,
                                                const void* gptr) {
    asm volatile("global_load_async_to_lds_b128 %0, %1, off"
                 :: "v"(lds_byte_off), "v"(gptr)
                 : "memory");
}
__device__ __forceinline__ void wait_async0() {
    asm volatile("s_wait_asynccnt 0" ::: "memory");
}

// ---------------------------------------------------------------------------
// A-fragment loader for V_WMMA_F32_16X16X32_BF16.
// ISA 16-bit A layout (16x32, MxK): lanes 0-15 -> M=lane, elems 0..7 = K 0..7,
// elems 8..15 = K 16..23; lanes 16-31 -> M=lane-16, elems = K 8..15 / 24..31.
// => two contiguous 8-bf16 (16B) loads per lane (ds_load_b128 from LDS).
// ---------------------------------------------------------------------------
__device__ __forceinline__ v16bf load_afrag(const __bf16* __restrict__ row_base_ptr,
                                            int half /*lane>=16*/) {
    const __bf16* r = row_base_ptr + half * 8;
    v8bf lo = *(const v8bf*)(r);
    v8bf hi = *(const v8bf*)(r + 16);
    v16bf a;
#pragma unroll
    for (int e = 0; e < 8; ++e) { a[e] = lo[e]; a[8 + e] = hi[e]; }
    return a;
}

// ---------------------------------------------------------------------------
// Pack W = [Whh | Wih] (G x (Hh+DIN), fp32 row-major) into B-fragment order:
// frag index = nt * nKt + kt ; inside a frag: lane*16 + e, contiguous 32B/lane.
// B[k][n] = W[n][k]; lane -> N=(lane&15), element e -> K as in A layout.
// Rows n >= realG are zero padding (used for the 49->64 padded NER head).
// ---------------------------------------------------------------------------
__global__ void pack_B_kernel(const float* __restrict__ Wih,
                              const float* __restrict__ Whh,
                              __bf16* __restrict__ outp,
                              int G, int Hh, int DIN, int realG) {
    long long K = Hh + DIN;
    long long total = (long long)G * K;
    long long idx = (long long)blockIdx.x * blockDim.x + threadIdx.x;
    if (idx >= total) return;
    int e = (int)(idx & 15);
    int lane = (int)((idx >> 4) & 31);
    long long frag = idx >> 9;
    int nKt = (int)(K / 32);
    int kt = (int)(frag % nKt);
    int nt = (int)(frag / nKt);
    int n = nt * 16 + (lane & 15);
    int half = (lane >> 4) & 1;
    int k = kt * 32 + e + (e >= 8 ? 8 : 0) + (half ? 8 : 0);
    float v = 0.0f;
    if (n < realG) {
        if (k < Hh) v = Whh[(long long)n * Hh + k];
        else        v = Wih[(long long)n * DIN + (k - Hh)];
    }
    outp[idx] = f2bf(v);
}

// ---------------------------------------------------------------------------
// Embedding gather -> bf16 activations x[(t*B+b)*E + e]
// ---------------------------------------------------------------------------
__global__ void embed_kernel(const float* __restrict__ emb,
                             const int* __restrict__ text,
                             __bf16* __restrict__ xout) {
    long long idx = (long long)blockIdx.x * blockDim.x + threadIdx.x;
    if (idx >= (long long)TT * BB * EE) return;
    int e = (int)(idx & (EE - 1));
    long long tb = idx >> 8;          // E==256
    int b = (int)(tb & (BB - 1));
    int t = (int)(tb >> 6);           // B==64
    int tok = text[b * TT + t];
    xout[idx] = f2bf(emb[(long long)tok * EE + e]);
}

// ---------------------------------------------------------------------------
// One bidirectional LSTM layer. Grid = 2 blocks (dir 0 = fwd, 1 = bwd),
// block = 1024 threads = 32 wave32 on one WGP. Templated on DIN so all K-loop
// bounds and B-fragment offsets are compile-time (single B pointer, gate
// offsets folded into the 24-bit global-load immediate; no spills).
// Gates = [h_t, x_t] @ [Whh|Wih]^T (bf16 WMMA, f32 accum); bias added in the
// pointwise phase so accumulators zero-init (rematerializable, spill-free).
// h (64x256 bf16) lives in LDS; x_t slab (64xDIN bf16) double-buffered in LDS,
// prefetched one step ahead with global_load_async_to_lds_b128 (ASYNCcnt).
// ---------------------------------------------------------------------------
template <int DIN>
__global__ void __launch_bounds__(1024)
lstm_bidir_kernel(const __bf16* __restrict__ x,       // (T*B, DIN) bf16
                  const __bf16* __restrict__ Bp_f,    // packed (1024 x (H+DIN))
                  const __bf16* __restrict__ Bp_b,
                  const float* __restrict__ bias_f,   // (4H)
                  const float* __restrict__ bias_b,
                  const int* __restrict__ text,       // (B, T)
                  __bf16* __restrict__ outh) {        // (T*B, 2H), dir half
    extern __shared__ char smem_raw[];
    __shared__ int mbuf[BB];

    constexpr int nKh = HH >> 5;          // h-region K tiles (8)
    constexpr int nKx = DIN >> 5;         // x-region K tiles
    constexpr int nKt = nKh + nKx;        // total K tiles
    constexpr unsigned slab = BB * DIN * 2;          // x slab bytes
    constexpr long long GSTRIDE = (long long)16 * nKt * 1024;  // bytes/gate

    const int dir = blockIdx.x;
    const __bf16* __restrict__ Bp = dir ? Bp_b : Bp_f;
    const float* __restrict__ bias = dir ? bias_b : bias_f;
    const int tid = threadIdx.x;
    const int w = tid >> 5;
    const int lane = tid & 31;
    const int half = (lane >> 4) & 1;
    const int lm = lane & 15;
    const int cg = w & 15;            // hidden column group (16 cols)
    const int rowbase = (w >> 4) * 32;
    const int m0 = rowbase + lm;      // A row, tile 0
    const int m1 = rowbase + 16 + lm; // A row, tile 1

    __bf16* hbuf = (__bf16*)smem_raw;                 // 64*256 bf16 = 32KB
    __bf16* xbufs = (__bf16*)(smem_raw + (size_t)BB * HH * 2);
    const unsigned xbase_off = lds_off_of(smem_raw) + (unsigned)(BB * HH * 2);

    for (int i = tid; i < BB * HH; i += 1024) hbuf[i] = f2bf(0.0f);

    // running global source pointer for async staging (advances +/- slab)
    const char* gs = (const char*)x + (dir ? (long long)(TT - 1) * slab : 0);

    // preload x slab for the first step into buffer 0
    for (unsigned off = (unsigned)tid * 16; off < slab; off += 1024u * 16u)
        async_copy_b128(xbase_off + off, gs + off);
    wait_async0();
    __syncthreads();

    float cst[2][8];
#pragma unroll
    for (int rt = 0; rt < 2; ++rt)
#pragma unroll
        for (int r = 0; r < 8; ++r) cst[rt][r] = 0.0f;

    float bgate[4];
#pragma unroll
    for (int g = 0; g < 4; ++g) bgate[g] = bias[g * HH + cg * 16 + lm];

    int cur = 0;
    for (int s = 0; s < TT; ++s) {
        const int t = dir ? (TT - 1 - s) : s;

        // mask for this step -> LDS (read after the mid-step barrier)
        if (tid < BB) mbuf[tid] = text[tid * TT + t];

        // prefetch next step's x slab into the other buffer (async, overlaps
        // this step's WMMA work; fenced before the end-of-step barrier)
        if (s + 1 < TT) {
            gs += dir ? -(long long)slab : (long long)slab;
            unsigned dbase = xbase_off + (unsigned)(1 - cur) * slab;
            for (unsigned off = (unsigned)tid * 16; off < slab;
                 off += 1024u * 16u)
                async_copy_b128(dbase + off, gs + off);
        }

        v8f acc[2][4];
#pragma unroll
        for (int rt = 0; rt < 2; ++rt)
#pragma unroll
            for (int g = 0; g < 4; ++g)
#pragma unroll
                for (int r = 0; r < 8; ++r) acc[rt][g][r] = 0.0f;

        // single B pointer; gate strides are compile-time immediates
        const char* bp = (const char*)Bp + (long long)cg * nKt * 1024 +
                         (long long)lane * 32;

        // --- h-region K tiles (A from hbuf) ---
#pragma unroll
        for (int kt = 0; kt < nKh; ++kt) {
            v16bf a0 = load_afrag(&hbuf[m0 * HH + kt * 32], half);
            v16bf a1 = load_afrag(&hbuf[m1 * HH + kt * 32], half);
#pragma unroll
            for (int g = 0; g < 4; ++g) {
                v16bf bfr = *(const v16bf*)(bp + (long long)kt * 1024 +
                                            g * GSTRIDE);
                acc[0][g] = __builtin_amdgcn_wmma_f32_16x16x32_bf16(
                    false, a0, false, bfr, (short)0, acc[0][g], false, false);
                acc[1][g] = __builtin_amdgcn_wmma_f32_16x16x32_bf16(
                    false, a1, false, bfr, (short)0, acc[1][g], false, false);
            }
        }
        // --- x-region K tiles (A from staged LDS slab) ---
        const __bf16* xb = xbufs + (size_t)cur * (BB * DIN);
#pragma unroll
        for (int kt = 0; kt < nKx; ++kt) {
            v16bf a0 = load_afrag(&xb[m0 * DIN + kt * 32], half);
            v16bf a1 = load_afrag(&xb[m1 * DIN + kt * 32], half);
#pragma unroll
            for (int g = 0; g < 4; ++g) {
                v16bf bfr = *(const v16bf*)(bp + (long long)(nKh + kt) * 1024 +
                                            g * GSTRIDE);
                acc[0][g] = __builtin_amdgcn_wmma_f32_16x16x32_bf16(
                    false, a0, false, bfr, (short)0, acc[0][g], false, false);
                acc[1][g] = __builtin_amdgcn_wmma_f32_16x16x32_bf16(
                    false, a1, false, bfr, (short)0, acc[1][g], false, false);
            }
        }
        __syncthreads();  // all waves done reading hbuf / mbuf ready

        // pointwise LSTM update; each (m, j) owned by exactly one lane
        const int j = cg * 16 + lm;
        __bf16* outrow = outh + (long long)t * BB * (2 * HH) + dir * HH + j;
#pragma unroll
        for (int rt = 0; rt < 2; ++rt) {
#pragma unroll
            for (int r = 0; r < 8; ++r) {
                int m = rowbase + rt * 16 + half * 8 + r;  // batch index
                float gi = acc[rt][0][r] + bgate[0];
                float gf = acc[rt][1][r] + bgate[1];
                float gc = acc[rt][2][r] + bgate[2];
                float go = acc[rt][3][r] + bgate[3];
                float c = cst[rt][r];
                float c2 = sigf(gf) * c + sigf(gi) * tanhf(gc);
                float h2 = sigf(go) * tanhf(c2);
                bool mv = (mbuf[m] != 0);
                float hold = bf2f(hbuf[m * HH + j]);
                float hn = mv ? h2 : hold;
                cst[rt][r] = mv ? c2 : c;
                __bf16 hb = f2bf(hn);
                hbuf[m * HH + j] = hb;
                outrow[(long long)m * (2 * HH)] = hb;
            }
        }
        wait_async0();    // next step's x slab fully landed in LDS
        __syncthreads();  // h writes + staged x visible to all waves
        cur ^= 1;
    }
}

// ---------------------------------------------------------------------------
// Emissions GEMM: em (32768 x 64pad) = h1 (32768 x 512) @ ner_pack + bias.
// One 16x16 C tile per wave; 512 blocks x 512 threads = 8192 waves.
// ---------------------------------------------------------------------------
__global__ void __launch_bounds__(512)
emis_kernel(const __bf16* __restrict__ X,   // (T*B, 512)
            const __bf16* __restrict__ Bp,  // packed (64 x 512)
            const float* __restrict__ bias, // (49)
            float* __restrict__ em) {       // (T*B, 64)
    const int gw = (int)((blockIdx.x * (long long)blockDim.x + threadIdx.x) >> 5);
    const int lane = threadIdx.x & 31;
    const int half = (lane >> 4) & 1;
    const int lm = lane & 15;
    const int nt = gw & 3;
    const int mt = gw >> 2;              // < 2048
    const int nKt = 512 >> 5;            // 16
    const int col = nt * 16 + lm;

    v8f acc;
#pragma unroll
    for (int r = 0; r < 8; ++r) acc[r] = 0.0f;

    const __bf16* bp = Bp + (((long long)nt * nKt) << 9) + lane * 16;
    const __bf16* src = X + ((long long)mt * 16 + lm) * 512;
#pragma unroll
    for (int kt = 0; kt < nKt; ++kt) {
        v16bf a = load_afrag(src + kt * 32, half);
        v16bf b = *(const v16bf*)(bp + kt * 512);
        acc = __builtin_amdgcn_wmma_f32_16x16x32_bf16(false, a, false, b,
                                                      (short)0, acc, false,
                                                      false);
    }
    float bv = (col < NTAG) ? bias[col] : 0.0f;
#pragma unroll
    for (int r = 0; r < 8; ++r) {
        int row = mt * 16 + half * 8 + r;
        em[(long long)row * 64 + col] = acc[r] + bv;
    }
}

// ---------------------------------------------------------------------------
// CRF NLL: numerator (raw tag chain, masked), forward logsumexp recursion
// (alpha double-buffered in LDS, exact max-subtracted), denominator, reduce.
// Single block, 1024 threads; batch rows independent across threads.
// ---------------------------------------------------------------------------
__global__ void __launch_bounds__(1024)
crf_kernel(const float* __restrict__ em,    // (T*B, 64)
           const int* __restrict__ text,    // (B, T)
           const int* __restrict__ ner,     // (B, T)
           const float* __restrict__ start,
           const float* __restrict__ endt,
           const float* __restrict__ trans, // (49, 49)
           float* __restrict__ out) {
    __shared__ float albuf[2][BB * 64];
    __shared__ float tr[NTAG * NTAG];
    __shared__ float red[1024];
    __shared__ float numb[BB];
    __shared__ float msum_sh;

    const int tid = threadIdx.x;

    for (int i = tid; i < NTAG * NTAG; i += 1024) tr[i] = trans[i];

    // mask count
    float cnt = 0.0f;
    for (int i = tid; i < BB * TT; i += 1024) cnt += (text[i] != 0) ? 1.0f : 0.0f;
    red[tid] = cnt;
    __syncthreads();
    if (tid == 0) {
        float s = 0.0f;
        for (int i = 0; i < 1024; ++i) s += red[i];
        msum_sh = s;
    }

    // numerator per batch row (raw consecutive-tag chain, masked terms only)
    if (tid < BB) {
        int b = tid;
        int tg0 = ner[b * TT + 0];
        float num = start[tg0] + em[(long long)(0 * BB + b) * 64 + tg0];
        int prev = tg0;
        int last = 0;
        for (int t = 1; t < TT; ++t) {
            int tg = ner[b * TT + t];
            if (text[b * TT + t] != 0) {
                num += tr[prev * NTAG + tg] +
                       em[((long long)t * BB + b) * 64 + tg];
                last = t;
            }
            prev = tg;
        }
        num += endt[ner[b * TT + last]];
        numb[b] = num;
    }

    // alpha init
    for (int s = tid; s < BB * NTAG; s += 1024) {
        int b = s / NTAG, j = s % NTAG;
        albuf[0][b * 64 + j] = start[j] + em[(long long)b * 64 + j];
    }

    int p = 0;
    for (int t = 1; t < TT; ++t) {
        __syncthreads();
        const float* cur = &albuf[p][0];
        float* nxt = &albuf[1 - p][0];
        for (int s = tid; s < BB * NTAG; s += 1024) {
            int b = s / NTAG, j = s % NTAG;
            if (text[b * TT + t] != 0) {
                float mx = -3.0e38f;
                for (int i = 0; i < NTAG; ++i) {
                    float v = cur[b * 64 + i] + tr[i * NTAG + j];
                    mx = fmaxf(mx, v);
                }
                float sum = 0.0f;
                for (int i = 0; i < NTAG; ++i)
                    sum += expf(cur[b * 64 + i] + tr[i * NTAG + j] - mx);
                nxt[b * 64 + j] =
                    mx + logf(sum) + em[((long long)t * BB + b) * 64 + j];
            } else {
                nxt[b * 64 + j] = cur[b * 64 + j];
            }
        }
        p ^= 1;
    }
    __syncthreads();

    if (tid < BB) {
        int b = tid;
        const float* al = &albuf[p][b * 64];
        float mx = -3.0e38f;
        for (int j = 0; j < NTAG; ++j) mx = fmaxf(mx, al[j] + endt[j]);
        float sum = 0.0f;
        for (int j = 0; j < NTAG; ++j) sum += expf(al[j] + endt[j] - mx);
        float den = mx + logf(sum);
        red[b] = numb[b] - den;
    }
    __syncthreads();
    if (tid == 0) {
        float s = 0.0f;
        for (int b = 0; b < BB; ++b) s += red[b];
        out[0] = -(s / msum_sh);
    }
}

// ---------------------------------------------------------------------------
extern "C" void kernel_launch(void* const* d_in, const int* in_sizes, int n_in,
                              void* d_out, int out_size, void* d_ws,
                              size_t ws_size, hipStream_t stream) {
    const float* emb     = (const float*)d_in[0];
    const float* l0f_Wih = (const float*)d_in[1];
    const float* l0f_Whh = (const float*)d_in[2];
    const float* l0f_b   = (const float*)d_in[3];
    const float* l0b_Wih = (const float*)d_in[4];
    const float* l0b_Whh = (const float*)d_in[5];
    const float* l0b_b   = (const float*)d_in[6];
    const float* l1f_Wih = (const float*)d_in[7];
    const float* l1f_Whh = (const float*)d_in[8];
    const float* l1f_b   = (const float*)d_in[9];
    const float* l1b_Wih = (const float*)d_in[10];
    const float* l1b_Whh = (const float*)d_in[11];
    const float* l1b_b   = (const float*)d_in[12];
    const float* ner_w   = (const float*)d_in[13];
    const float* ner_b   = (const float*)d_in[14];
    const float* start_t = (const float*)d_in[15];
    const float* end_t   = (const float*)d_in[16];
    const float* trans   = (const float*)d_in[17];
    const int*   text    = (const int*)d_in[18];
    const int*   ner     = (const int*)d_in[19];
    float* out = (float*)d_out;

    // workspace partition (256B aligned); total ~= 98 MB
    char* p = (char*)d_ws;
    auto alloc = [&](size_t bytes) -> void* {
        void* r = (void*)p;
        p += (bytes + 255) & ~(size_t)255;
        return r;
    };
    __bf16* x_bf   = (__bf16*)alloc((size_t)TT * BB * EE * 2);        // 16.8MB
    __bf16* h0_bf  = (__bf16*)alloc((size_t)TT * BB * 2 * HH * 2);    // 33.6MB
    __bf16* h1_bf  = (__bf16*)alloc((size_t)TT * BB * 2 * HH * 2);    // 33.6MB
    float*  em     = (float*)alloc((size_t)TT * BB * 64 * 4);         // 8.4MB
    __bf16* pk0f   = (__bf16*)alloc((size_t)1024 * 512 * 2);
    __bf16* pk0b   = (__bf16*)alloc((size_t)1024 * 512 * 2);
    __bf16* pk1f   = (__bf16*)alloc((size_t)1024 * 768 * 2);
    __bf16* pk1b   = (__bf16*)alloc((size_t)1024 * 768 * 2);
    __bf16* pkner  = (__bf16*)alloc((size_t)64 * 512 * 2);

    // 1. pack weights into WMMA B-fragment layout (bf16)
    {
        long long n0 = (long long)1024 * 512;
        pack_B_kernel<<<dim3((unsigned)((n0 + 255) / 256)), 256, 0, stream>>>(
            l0f_Wih, l0f_Whh, pk0f, 1024, HH, EE, 1024);
        pack_B_kernel<<<dim3((unsigned)((n0 + 255) / 256)), 256, 0, stream>>>(
            l0b_Wih, l0b_Whh, pk0b, 1024, HH, EE, 1024);
        long long n1 = (long long)1024 * 768;
        pack_B_kernel<<<dim3((unsigned)((n1 + 255) / 256)), 256, 0, stream>>>(
            l1f_Wih, l1f_Whh, pk1f, 1024, HH, 2 * HH, 1024);
        pack_B_kernel<<<dim3((unsigned)((n1 + 255) / 256)), 256, 0, stream>>>(
            l1b_Wih, l1b_Whh, pk1b, 1024, HH, 2 * HH, 1024);
        long long nn = (long long)64 * 512;
        pack_B_kernel<<<dim3((unsigned)((nn + 255) / 256)), 256, 0, stream>>>(
            ner_w, nullptr, pkner, 64, 0, 2 * HH, NTAG);
    }

    // 2. embedding gather -> bf16
    {
        long long n = (long long)TT * BB * EE;
        embed_kernel<<<dim3((unsigned)((n + 255) / 256)), 256, 0, stream>>>(
            emb, text, x_bf);
    }

    // 3. layer 0 (fwd+bwd concurrently): LDS = 32KB h + 2*32KB x slabs
    {
        size_t shmem = (size_t)BB * HH * 2 + 2 * (size_t)BB * EE * 2; // 96KB
        lstm_bidir_kernel<EE><<<2, 1024, shmem, stream>>>(
            x_bf, pk0f, pk0b, l0f_b, l0b_b, text, h0_bf);
    }
    // 4. layer 1: LDS = 32KB h + 2*64KB x slabs = 160KB (< 320KB/WGP)
    {
        size_t shmem = (size_t)BB * HH * 2 + 2 * (size_t)BB * 2 * HH * 2;
        lstm_bidir_kernel<2 * HH><<<2, 1024, shmem, stream>>>(
            h0_bf, pk1f, pk1b, l1f_b, l1b_b, text, h1_bf);
    }

    // 5. emissions
    emis_kernel<<<512, 512, 0, stream>>>(h1_bf, pkner, ner_b, em);

    // 6. CRF NLL -> scalar
    crf_kernel<<<1, 1024, 0, stream>>>(em, text, ner, start_t, end_t, trans,
                                       out);
}
